// FactorGraphDecoderV0_86260123173232
// MI455X (gfx1250) — matrix operand order repaired and checked by reading, not compiled, not run on hardware
//
#include <hip/hip_runtime.h>
#include <hip/hip_bf16.h>
#include <cstdint>
#include <cstddef>

// ---------------------------------------------------------------------------
// Problem constants (from reference)
// ---------------------------------------------------------------------------
#define BB 4
#define ND 20000
#define NE 40000
#define EE 150000
#define HH 128
#define KK 3
#define LN_EPS 1e-5f

typedef __attribute__((ext_vector_type(16))) __bf16 v16bf;
typedef __attribute__((ext_vector_type(8)))  float  v8f;
typedef unsigned int u32x4 __attribute__((ext_vector_type(4)));
typedef int          i32x4 __attribute__((ext_vector_type(4)));
typedef int          i32x8 __attribute__((ext_vector_type(8)));

// ---------------------------------------------------------------------------
// Small utility kernels
// ---------------------------------------------------------------------------
__global__ void zero_f32(float* __restrict__ p, long n) {
    long i = (long)blockIdx.x * blockDim.x + threadIdx.x;
    long stride = (long)gridDim.x * blockDim.x;
    for (; i < n; i += stride) p[i] = 0.0f;
}

// Pre-swizzle a stack of (nmat, H, H) f32 weights into WMMA B-fragment order:
//   fragment f = kk*8 + nt; lane l holds 16 contiguous bf16:
//     element i  <- W[kk*32 + (l>>4)*16 + i][nt*16 + (l&15)]
__global__ void swizzle_w(const float* __restrict__ W, __bf16* __restrict__ Wf, int nmat) {
    long total = (long)nmat * HH * HH;
    long g = (long)blockIdx.x * blockDim.x + threadIdx.x;
    long stride = (long)gridDim.x * blockDim.x;
    for (; g < total; g += stride) {
        int mat = (int)(g >> 14);
        int idx = (int)(g & 16383);
        int f   = idx >> 9;
        int rem = idx & 511;
        int l   = rem >> 4;
        int i   = rem & 15;
        int kk  = f >> 3;
        int nt  = f & 7;
        int krow = kk * 32 + ((l >> 4) << 4) + i;
        int col  = nt * 16 + (l & 15);
        Wf[g] = (__bf16)W[(long)mat * HH * HH + krow * HH + col];
    }
}

// Edge prep: counts (scatter degree) + per-edge sigmoid weight
__global__ void prep_edges(const int* __restrict__ e_dst, const int* __restrict__ d_dst,
                           const float* __restrict__ err_w,
                           float* __restrict__ cnt_e, float* __restrict__ cnt_d,
                           float* __restrict__ wE) {
    int i = blockIdx.x * blockDim.x + threadIdx.x;
    int stride = gridDim.x * blockDim.x;
    for (; i < EE; i += stride) {
        int ed = e_dst[i];
        atomicAdd(&cnt_e[ed], 1.0f);
        atomicAdd(&cnt_d[d_dst[i]], 1.0f);
        float w = err_w[ed];
        wE[i] = 1.0f / (1.0f + __expf(-w));
    }
}

__global__ void mask_count(const unsigned char* __restrict__ mask, float* __restrict__ cnt) {
    int i = blockIdx.x * blockDim.x + threadIdx.x;
    int stride = gridDim.x * blockDim.x;
    for (; i < NE; i += stride)
        if (mask[i]) atomicAdd(cnt, 1.0f);
}

// Input projections: outer product + bias + relu; writes f32 and bf16 copies.
__global__ void proj_det(const float* __restrict__ x, const float* __restrict__ W,
                         const float* __restrict__ b, float* __restrict__ h,
                         __bf16* __restrict__ hb) {
    long total = (long)BB * ND * HH;
    long i = (long)blockIdx.x * blockDim.x + threadIdx.x;
    long stride = (long)gridDim.x * blockDim.x;
    for (; i < total; i += stride) {
        int c = (int)(i & (HH - 1));
        long row = i >> 7;
        float v = x[row] * W[c] + b[c];
        v = v > 0.0f ? v : 0.0f;
        h[i] = v; hb[i] = (__bf16)v;
    }
}

__global__ void proj_err(const float* __restrict__ x, const float* __restrict__ W,
                         const float* __restrict__ b, float* __restrict__ h,
                         __bf16* __restrict__ hb) {
    long total = (long)BB * NE * HH;
    long i = (long)blockIdx.x * blockDim.x + threadIdx.x;
    long stride = (long)gridDim.x * blockDim.x;
    for (; i < total; i += stride) {
        int c = (int)(i & (HH - 1));
        long row = i >> 7;
        int n = (int)(row % NE);
        float v = x[n] * W[c] + b[c];
        v = v > 0.0f ? v : 0.0f;
        h[i] = v; hb[i] = (__bf16)v;
    }
}

// ---------------------------------------------------------------------------
// Weight staging: Tensor Data Mover (TDM) if available, else vector copy.
// One 1-row tile of 16384 x 2-byte elements, global -> LDS, wave 0 issues.
// ---------------------------------------------------------------------------
__device__ __forceinline__ void stage_weights(const __bf16* __restrict__ Wf,
                                              __bf16* sW, int tid) {
#if defined(__gfx1250__) && __has_builtin(__builtin_amdgcn_tensor_load_to_lds)
    if (tid < 32) {                           // wave 0 only (EXEC all-ones)
        unsigned long long ga = (unsigned long long)(uintptr_t)Wf;
        u32x4 g0;
        g0[0] = 1u;                                        // count=1, user mode
        g0[1] = (unsigned)(uintptr_t)sW;                   // lds_addr
        g0[2] = (unsigned)(ga & 0xFFFFFFFFull);            // global_addr lo
        g0[3] = (unsigned)((ga >> 32) & 0x01FFFFFFull)     // global_addr[56:32]
                | (2u << 30);                              // type = 2 (image)
        i32x8 g1;
        g1[0] = (int)(1u << 16);        // data_size=1 (2 bytes), mask=0
        g1[1] = (int)(16384u << 16);    // tensor_dim0[15:0] at bits 63:48
        g1[2] = (int)(1u << 16);        // tensor_dim0 hi=0; tensor_dim1=1
        g1[3] = (int)(16384u << 16);    // tile_dim0 = 16384 at bits 127:112
        g1[4] = 1;                      // tile_dim1 = 1
        g1[5] = 16384;                  // tensor_dim0_stride lo32
        g1[6] = 0;
        g1[7] = 0;
        i32x4 gz = {0, 0, 0, 0};
#if __clang_major__ >= 23
        i32x8 gz8 = {0, 0, 0, 0, 0, 0, 0, 0};
        __builtin_amdgcn_tensor_load_to_lds(g0, g1, gz, gz, gz8, 0);
#else
        __builtin_amdgcn_tensor_load_to_lds(g0, g1, gz, gz, 0);
#endif
        __builtin_amdgcn_s_wait_tensorcnt(0);
    }
#else
    const uint4* wsrc = (const uint4*)Wf;
    uint4* wdst = (uint4*)sW;
    for (int i = tid; i < (HH * HH * 2) / 16; i += 128) wdst[i] = wsrc[i];
#endif
}

// ---------------------------------------------------------------------------
// WMMA fragment helpers: all loads are 16-byte LDS loads (ds_load_b128)
// ---------------------------------------------------------------------------
union frag_cv { uint4 u[2]; v16bf v; };

__device__ __forceinline__ v16bf fragA(const __bf16* sA, int lane, int kk) {
    int m  = lane & 15;
    int kb = (lane >> 4) << 4;
    const __bf16* r = sA + m * HH + kk * 32;
    frag_cv cv;
    cv.u[0] = *(const uint4*)(r + kb);
    cv.u[1] = *(const uint4*)(r + kb + 8);
    return cv.v;
}

__device__ __forceinline__ v16bf fragB(const __bf16* sWf, int lane, int kk, int nt) {
    const __bf16* p = sWf + ((kk * 8 + nt) << 9) + (lane << 4);
    frag_cv cv;
    cv.u[0] = *(const uint4*)(p);
    cv.u[1] = *(const uint4*)(p + 8);
    return cv.v;
}

// ---------------------------------------------------------------------------
// Edge message kernel:  agg[b, dst[e], :] += wE[e] * (hSrc[b, src[e], :] @ W)
// M=32 edges per tile: two A row-tiles share each B fragment (register reuse),
// 16 WMMAs per wave per tile. Uniform base + 32-bit offsets (saddr form).
// ---------------------------------------------------------------------------
#define EM 32   // edges per tile

__global__ void __launch_bounds__(128)
edge_msg_wmma(const __bf16* __restrict__ hSrc,   // (B, Nsrc, H) bf16
              const __bf16* __restrict__ Wf,     // (H,H) bf16 pre-swizzled
              const int* __restrict__ src_idx,   // (E,)
              const int* __restrict__ dst_idx,   // (E,)
              const float* __restrict__ wE,      // (E,) or nullptr
              float* __restrict__ agg,           // (B, Ndst, H) f32
              int Nsrc, int Ndst) {
    __shared__ __bf16 sW[HH * HH];      // 32 KB, fragment order
    __shared__ __bf16 sA[EM * HH];      // 8 KB
    __shared__ unsigned sSrcOff[EM];    // element offsets (row * H)
    __shared__ unsigned sDstOff[EM];
    __shared__ float    sWe[EM];

    const int tid  = threadIdx.x;
    const int lane = tid & 31;
    const int wv   = tid >> 5;
    const int nt0  = wv * 2;
    const int ncol = wv * 32;
    const int b    = blockIdx.y;

    const __bf16* hSrcB = hSrc + (size_t)b * Nsrc * HH;   // uniform base
    float*        aggB  = agg  + (size_t)b * Ndst * HH;   // uniform base

    stage_weights(Wf, sW, tid);
    __syncthreads();

    const int ntiles = (EE + EM - 1) / EM;
    for (int tile = blockIdx.x; tile < ntiles; tile += gridDim.x) {
        const int e0 = tile * EM;
        if (tid < EM) {
            int e = e0 + tid;
            int ok = (e < EE);
            sSrcOff[tid] = (ok ? (unsigned)src_idx[e] : 0u) * HH;
            sDstOff[tid] = (ok ? (unsigned)dst_idx[e] : 0u) * HH;
            sWe[tid]     = (ok && wE) ? wE[e] : 1.0f;
        }
        int nxt = tile + gridDim.x;
        if (nxt < ntiles) __builtin_prefetch(src_idx + nxt * EM, 0, 1);
        __syncthreads();

        // gather EM source rows: thread t -> rows t/8 and 16+t/8, 32B seg t%8
        {
            int r = tid >> 3, seg = tid & 7;
#pragma unroll
            for (int t = 0; t < 2; ++t) {
                int rr = r + t * 16;
                const uint4* p = (const uint4*)(hSrcB + sSrcOff[rr] + seg * 16);
                uint4 u0 = p[0], u1 = p[1];
                uint4* q = (uint4*)(sA + rr * HH + seg * 16);
                q[0] = u0; q[1] = u1;
            }
        }
        __syncthreads();

        v8f acc00 = {}, acc01 = {};   // rows 0-15  x (ncol, ncol+16)
        v8f acc10 = {}, acc11 = {};   // rows 16-31 x (ncol, ncol+16)
#pragma unroll
        for (int kk = 0; kk < 4; ++kk) {
            v16bf a0 = fragA(sA,            lane, kk);
            v16bf a1 = fragA(sA + 16 * HH,  lane, kk);
            v16bf b0 = fragB(sW, lane, kk, nt0);
            v16bf b1 = fragB(sW, lane, kk, nt0 + 1);
            acc00 = __builtin_amdgcn_wmma_f32_16x16x32_bf16(false, a0, false, b0,
                                                            (short)0, acc00, false, false);
            acc01 = __builtin_amdgcn_wmma_f32_16x16x32_bf16(false, a0, false, b1,
                                                            (short)0, acc01, false, false);
            acc10 = __builtin_amdgcn_wmma_f32_16x16x32_bf16(false, a1, false, b0,
                                                            (short)0, acc10, false, false);
            acc11 = __builtin_amdgcn_wmma_f32_16x16x32_bf16(false, a1, false, b1,
                                                            (short)0, acc11, false, false);
        }

        // scatter: C layout -> element v at (M = v + 8*(lane/16), N = lane&15)
#pragma unroll
        for (int v = 0; v < 8; ++v) {
            int m0 = v + ((lane >> 4) << 3);
            unsigned col = (unsigned)(ncol + (lane & 15));
            {
                int e = e0 + m0;
                if (e < EE) {
                    float sc = sWe[m0];
                    unsigned off = sDstOff[m0] + col;
                    atomicAdd(aggB + off,      acc00[v] * sc);
                    atomicAdd(aggB + off + 16, acc01[v] * sc);
                }
            }
            {
                int m1 = m0 + 16;
                int e = e0 + m1;
                if (e < EE) {
                    float sc = sWe[m1];
                    unsigned off = sDstOff[m1] + col;
                    atomicAdd(aggB + off,      acc10[v] * sc);
                    atomicAdd(aggB + off + 16, acc11[v] * sc);
                }
            }
        }
        __syncthreads();
    }
}

// ---------------------------------------------------------------------------
// Node update:  hX = LN(hX + relu(hX @ Wself + agg/cnt + bias); gamma, beta)
// ---------------------------------------------------------------------------
__global__ void __launch_bounds__(128)
node_update_wmma(float* __restrict__ hX,             // (B,N,H) f32 in/out
                 __bf16* __restrict__ hXb,           // (B,N,H) bf16 in/out
                 const __bf16* __restrict__ Wf,      // (H,H) pre-swizzled
                 const float* __restrict__ agg,      // (B,N,H)
                 const float* __restrict__ cnt,      // (N,)
                 const float* __restrict__ bias,     // (H,)
                 const float* __restrict__ gamma,    // (H,)
                 const float* __restrict__ beta,     // (H,)
                 int N) {
    __shared__ __bf16 sW[HH * HH];      // 32 KB
    __shared__ __bf16 sA[16 * HH];      // 4 KB
    __shared__ float  sOut[16 * HH];    // 8 KB
    __shared__ float  sP1[128], sP2[128];
    __shared__ float  sMu[16], sRs[16];

    const int tid  = threadIdx.x;
    const int lane = tid & 31;
    const int wv   = tid >> 5;
    const int nt0  = wv * 2;
    const int ncol = wv * 32;
    const int b    = blockIdx.y;

    float*        hXB  = hX  + (size_t)b * N * HH;   // uniform bases
    __bf16*       hXbB = hXb + (size_t)b * N * HH;
    const float*  aggB = agg + (size_t)b * N * HH;

    stage_weights(Wf, sW, tid);
    __syncthreads();

    const int ntiles = N / 16;
    for (int tile = blockIdx.x; tile < ntiles; tile += gridDim.x) {
        const unsigned n0 = tile * 16;
        {
            int r = tid >> 3, seg = tid & 7;
            const uint4* p = (const uint4*)(hXbB + (n0 + r) * HH + seg * 16);
            uint4 u0 = p[0], u1 = p[1];
            uint4* q = (uint4*)(sA + r * HH + seg * 16);
            q[0] = u0; q[1] = u1;
        }
        __syncthreads();

        v8f acc0 = {}, acc1 = {};
#pragma unroll
        for (int kk = 0; kk < 4; ++kk) {
            v16bf a  = fragA(sA, lane, kk);
            v16bf b0 = fragB(sW, lane, kk, nt0);
            v16bf b1 = fragB(sW, lane, kk, nt0 + 1);
            acc0 = __builtin_amdgcn_wmma_f32_16x16x32_bf16(false, a, false, b0,
                                                           (short)0, acc0, false, false);
            acc1 = __builtin_amdgcn_wmma_f32_16x16x32_bf16(false, a, false, b1,
                                                           (short)0, acc1, false, false);
        }

        // fused: + agg/cnt + bias -> relu -> + residual, stash in LDS
#pragma unroll
        for (int v = 0; v < 8; ++v) {
            int m = v + ((lane >> 4) << 3);
            unsigned nidx = n0 + m;
            unsigned base = nidx * HH;
            float cn = fmaxf(cnt[nidx], 1.0f);
            float rcn = 1.0f / cn;
            int c0 = ncol + (lane & 15);
            float p0 = acc0[v] + aggB[base + c0] * rcn + bias[c0];
            float p1 = acc1[v] + aggB[base + c0 + 16] * rcn + bias[c0 + 16];
            p0 = p0 > 0.0f ? p0 : 0.0f;
            p1 = p1 > 0.0f ? p1 : 0.0f;
            sOut[m * HH + c0]      = hXB[base + c0] + p0;
            sOut[m * HH + c0 + 16] = hXB[base + c0 + 16] + p1;
        }
        __syncthreads();

        // LayerNorm over H=128 per row
        int r = tid >> 3, seg = tid & 7;
        float s1 = 0.0f, s2 = 0.0f;
#pragma unroll
        for (int i = 0; i < 16; ++i) {
            float x = sOut[r * HH + seg * 16 + i];
            s1 += x; s2 += x * x;
        }
        sP1[tid] = s1; sP2[tid] = s2;
        __syncthreads();
        if (tid < 16) {
            float a = 0.0f, q = 0.0f;
#pragma unroll
            for (int j = 0; j < 8; ++j) { a += sP1[tid * 8 + j]; q += sP2[tid * 8 + j]; }
            float mu = a * (1.0f / HH);
            float var = q * (1.0f / HH) - mu * mu;
            sMu[tid] = mu;
            sRs[tid] = rsqrtf(var + LN_EPS);
        }
        __syncthreads();
        {
            unsigned base = (n0 + r) * HH;
            float mu = sMu[r], rs = sRs[r];
#pragma unroll
            for (int i = 0; i < 16; i += 2) {
                int c = seg * 16 + i;
                float y0 = gamma[c]     * (sOut[r * HH + c]     - mu) * rs + beta[c];
                float y1 = gamma[c + 1] * (sOut[r * HH + c + 1] - mu) * rs + beta[c + 1];
                *(float2*)&hXB[base + c] = make_float2(y0, y1);
                __bf16 t0 = (__bf16)y0, t1 = (__bf16)y1;
                uint u = (uint)(*(unsigned short*)&t0) | ((uint)(*(unsigned short*)&t1) << 16);
                *(uint*)&hXbB[base + c] = u;
            }
        }
        __syncthreads();
    }
}

// ---------------------------------------------------------------------------
// Readout: masked mean + max over hE -> emb (B, 2H) -> MLP head -> (B,1)
// ---------------------------------------------------------------------------
#define RCHUNKS 64

__global__ void readout_stage1(const float* __restrict__ hE,
                               const unsigned char* __restrict__ mask,
                               float* __restrict__ psum, float* __restrict__ pmax) {
    int c = blockIdx.x;
    int b = blockIdx.y;
    int h = threadIdx.x;
    int chunk = (NE + RCHUNKS - 1) / RCHUNKS;
    int n0 = c * chunk;
    int n1 = min(NE, n0 + chunk);
    const float* hEB = hE + (size_t)b * NE * HH;
    float s = 0.0f, mx = -INFINITY;
    for (int n = n0; n < n1; ++n) {
        if (mask[n]) {
            float v = hEB[(unsigned)n * HH + h];
            s += v;
            mx = fmaxf(mx, v);
        }
    }
    psum[((size_t)b * RCHUNKS + c) * HH + h] = s;
    pmax[((size_t)b * RCHUNKS + c) * HH + h] = mx;
}

__global__ void readout_stage2(const float* __restrict__ psum,
                               const float* __restrict__ pmax,
                               const float* __restrict__ mcnt,
                               float* __restrict__ emb) {
    int b = blockIdx.x;
    int h = threadIdx.x;
    float s = 0.0f, mx = -INFINITY;
    for (int c = 0; c < RCHUNKS; ++c) {
        s += psum[((size_t)b * RCHUNKS + c) * HH + h];
        mx = fmaxf(mx, pmax[((size_t)b * RCHUNKS + c) * HH + h]);
    }
    float cn = fmaxf(mcnt[0], 1.0f);
    emb[b * 2 * HH + h]      = s / cn;
    emb[b * 2 * HH + HH + h] = mx;
}

__global__ void head_mlp(const float* __restrict__ emb,
                         const float* __restrict__ W1, const float* __restrict__ b1,
                         const float* __restrict__ W2, const float* __restrict__ b2,
                         float* __restrict__ out) {
    __shared__ float sRed[128];
    int b = blockIdx.x;
    int h = threadIdx.x;
    float acc = b1[h];
    for (int j = 0; j < 2 * HH; ++j)
        acc += emb[b * 2 * HH + j] * W1[j * HH + h];
    acc = acc > 0.0f ? acc : 0.0f;
    sRed[h] = acc * W2[h];
    __syncthreads();
    for (int off = 64; off > 0; off >>= 1) {
        if (h < off) sRed[h] += sRed[h + off];
        __syncthreads();
    }
    if (h == 0) out[b] = sRed[0] + b2[0];
}

// ---------------------------------------------------------------------------
// Host-side launch
// ---------------------------------------------------------------------------
static inline char* ws_align(char* p) {
    return (char*)(((uintptr_t)p + 255) & ~(uintptr_t)255);
}

extern "C" void kernel_launch(void* const* d_in, const int* in_sizes, int n_in,
                              void* d_out, int out_size, void* d_ws, size_t ws_size,
                              hipStream_t stream) {
    const float* det_features = (const float*)d_in[0];
    const float* err_features = (const float*)d_in[1];
    const int*   d_src        = (const int*)d_in[2];
    const int*   e_dst        = (const int*)d_in[3];
    const int*   e_src        = (const int*)d_in[4];
    const int*   d_dst        = (const int*)d_in[5];
    const float* error_w      = (const float*)d_in[6];
    const unsigned char* mask = (const unsigned char*)d_in[7];
    const float* det_W = (const float*)d_in[8];
    const float* det_b = (const float*)d_in[9];
    const float* err_W = (const float*)d_in[10];
    const float* err_b = (const float*)d_in[11];
    const float* W_d2e    = (const float*)d_in[12];
    const float* W_e_self = (const float*)d_in[13];
    const float* b_e      = (const float*)d_in[14];
    const float* gamma_e  = (const float*)d_in[15];
    const float* beta_e   = (const float*)d_in[16];
    const float* W_e2d    = (const float*)d_in[17];
    const float* W_d_self = (const float*)d_in[18];
    const float* b_d      = (const float*)d_in[19];
    const float* gamma_d  = (const float*)d_in[20];
    const float* beta_d   = (const float*)d_in[21];
    const float* head_W1  = (const float*)d_in[22];
    const float* head_b1  = (const float*)d_in[23];
    const float* head_W2  = (const float*)d_in[24];
    const float* head_b2  = (const float*)d_in[25];
    float* out = (float*)d_out;

    // ---- workspace carve-up ----
    char* p = (char*)d_ws;
    float*  hD   = (float*)p;            p = ws_align(p + (size_t)BB * ND * HH * 4);
    float*  hE   = (float*)p;            p = ws_align(p + (size_t)BB * NE * HH * 4);
    __bf16* hDb  = (__bf16*)p;           p = ws_align(p + (size_t)BB * ND * HH * 2);
    __bf16* hEb  = (__bf16*)p;           p = ws_align(p + (size_t)BB * NE * HH * 2);
    float*  aggE = (float*)p;            p = ws_align(p + (size_t)BB * NE * HH * 4);
    float*  aggD = (float*)p;            p = ws_align(p + (size_t)BB * ND * HH * 4);
    __bf16* Wd2e_b = (__bf16*)p;         p = ws_align(p + (size_t)KK * HH * HH * 2);
    __bf16* Wes_b  = (__bf16*)p;         p = ws_align(p + (size_t)KK * HH * HH * 2);
    __bf16* We2d_b = (__bf16*)p;         p = ws_align(p + (size_t)KK * HH * HH * 2);
    __bf16* Wds_b  = (__bf16*)p;         p = ws_align(p + (size_t)KK * HH * HH * 2);
    float* cnt_e = (float*)p;            p = ws_align(p + (size_t)NE * 4);
    float* cnt_d = (float*)p;            p = ws_align(p + (size_t)ND * 4);
    float* wE    = (float*)p;            p = ws_align(p + (size_t)EE * 4);
    float* mcnt  = (float*)p;            p = ws_align(p + 256);
    float* psum  = (float*)p;            p = ws_align(p + (size_t)BB * RCHUNKS * HH * 4);
    float* pmax  = (float*)p;            p = ws_align(p + (size_t)BB * RCHUNKS * HH * 4);
    float* emb   = (float*)p;            p = ws_align(p + (size_t)BB * 2 * HH * 4);
    (void)ws_size; (void)in_sizes; (void)n_in; (void)out_size;

    // ---- prep ----
    zero_f32<<<256, 256, 0, stream>>>(cnt_e, (long)NE);
    zero_f32<<<256, 256, 0, stream>>>(cnt_d, (long)ND);
    zero_f32<<<1, 64, 0, stream>>>(mcnt, 1);
    prep_edges<<<512, 256, 0, stream>>>(e_dst, d_dst, error_w, cnt_e, cnt_d, wE);
    mask_count<<<256, 256, 0, stream>>>(mask, mcnt);

    swizzle_w<<<192, 256, 0, stream>>>(W_d2e,    Wd2e_b, KK);
    swizzle_w<<<192, 256, 0, stream>>>(W_e_self, Wes_b,  KK);
    swizzle_w<<<192, 256, 0, stream>>>(W_e2d,    We2d_b, KK);
    swizzle_w<<<192, 256, 0, stream>>>(W_d_self, Wds_b,  KK);

    proj_det<<<2048, 256, 0, stream>>>(det_features, det_W, det_b, hD, hDb);
    proj_err<<<2048, 256, 0, stream>>>(err_features, err_W, err_b, hE, hEb);

    // ---- K message-passing layers ----
    dim3 egrid(1024, BB), eblk(128);
    dim3 ngrid(1024, BB), nblk(128);
    for (int k = 0; k < KK; ++k) {
        const __bf16* wd2e = Wd2e_b + (size_t)k * HH * HH;
        const __bf16* wes  = Wes_b  + (size_t)k * HH * HH;
        const __bf16* we2d = We2d_b + (size_t)k * HH * HH;
        const __bf16* wds  = Wds_b  + (size_t)k * HH * HH;

        // D -> E
        zero_f32<<<2048, 256, 0, stream>>>(aggE, (long)BB * NE * HH);
        edge_msg_wmma<<<egrid, eblk, 0, stream>>>(hDb, wd2e, d_src, e_dst, wE, aggE, ND, NE);
        node_update_wmma<<<ngrid, nblk, 0, stream>>>(hE, hEb, wes, aggE, cnt_e,
                                                     b_e + k * HH, gamma_e + k * HH,
                                                     beta_e + k * HH, NE);
        // E -> D
        zero_f32<<<2048, 256, 0, stream>>>(aggD, (long)BB * ND * HH);
        edge_msg_wmma<<<egrid, eblk, 0, stream>>>(hEb, we2d, e_src, d_dst, nullptr, aggD, NE, ND);
        node_update_wmma<<<ngrid, nblk, 0, stream>>>(hD, hDb, wds, aggD, cnt_d,
                                                     b_d + k * HH, gamma_d + k * HH,
                                                     beta_d + k * HH, ND);
    }

    // ---- readout + head ----
    readout_stage1<<<dim3(RCHUNKS, BB), 128, 0, stream>>>(hE, mask, psum, pmax);
    readout_stage2<<<BB, 128, 0, stream>>>(psum, pmax, mcnt, emb);
    head_mlp<<<BB, 128, 0, stream>>>(emb, head_W1, head_b1, head_W2, head_b2, out);
}
